// TransformerBlock_61229053772256
// MI455X (gfx1250) — matrix operand order repaired and checked by reading, not compiled, not run on hardware
//
#include <hip/hip_runtime.h>
#include <cstdint>
#include <cstddef>

// ---------------------------------------------------------------------------
// Transformer block for MI455X (gfx1250, wave32, WMMA).
// All GEMMs: bf16 inputs, f32 accumulation via v_wmma_f32_16x16x32_bf16.
// Staging: double-buffered async global->LDS (ASYNCcnt) when available.
// All B operands are consumed in [N][K] row-major ("B^T source") form:
//   - weights are transposed+converted once up front,
//   - K (for Q@K^T) is naturally [N][K],
//   - V is produced transposed by a dedicated GEMM epilogue.
// ---------------------------------------------------------------------------

#define DIM   1024
#define SEQ   2048
#define BATCH 4
#define FF    2048
#define ROWS  (BATCH * SEQ)   // 8192
#define LDSK  40              // padded LDS row stride (bf16): 32 data + 8 pad

typedef __attribute__((ext_vector_type(16))) __bf16 v16bf;
typedef __attribute__((ext_vector_type(8)))  float  v8f;
typedef int v4i_async __attribute__((vector_size(16)));

#if __has_builtin(__builtin_amdgcn_global_load_async_to_lds_b128)
#define ASYNC_LDS 1
typedef __attribute__((address_space(1))) v4i_async* gas_ptr;
typedef __attribute__((address_space(3))) v4i_async* las_ptr;
#endif

__device__ __forceinline__ unsigned short f2bf(float f) {
    union { float f; uint32_t u; } v; v.f = f;
    uint32_t r = v.u + 0x7FFFu + ((v.u >> 16) & 1u);   // round-to-nearest-even
    return (unsigned short)(r >> 16);
}

// Copy 16 bf16 (32 bytes) global -> LDS. Async path issues 2 b128 ops
// (ASYNCcnt += 2 per call; a full A+B stage = 4 per thread).
__device__ __forceinline__ void stage32B(unsigned short* lds,
                                         const unsigned short* g) {
#ifdef ASYNC_LDS
    __builtin_amdgcn_global_load_async_to_lds_b128(
        (gas_ptr)(g), (las_ptr)(lds), 0, 0);
    __builtin_amdgcn_global_load_async_to_lds_b128(
        (gas_ptr)(g + 8), (las_ptr)(lds + 8), 0, 0);
#else
    ((uint4*)lds)[0] = ((const uint4*)g)[0];
    ((uint4*)lds)[1] = ((const uint4*)g)[1];
#endif
}

__device__ __forceinline__ void wait_async_4() {
#ifdef ASYNC_LDS
#if __has_builtin(__builtin_amdgcn_s_wait_asynccnt)
    __builtin_amdgcn_s_wait_asynccnt(4);
#else
    asm volatile("s_wait_asynccnt 0x4" ::: "memory");
#endif
#endif
}
__device__ __forceinline__ void wait_async_0() {
#ifdef ASYNC_LDS
#if __has_builtin(__builtin_amdgcn_s_wait_asynccnt)
    __builtin_amdgcn_s_wait_asynccnt(0);
#else
    asm volatile("s_wait_asynccnt 0x0" ::: "memory");
#endif
#endif
}

// ---------------------------------------------------------------------------
// Transpose + fp32->bf16 convert: in[R][C] fp32 -> out[C][R] bf16.
// ---------------------------------------------------------------------------
__global__ __launch_bounds__(256)
void transpose_cvt(const float* __restrict__ in, unsigned short* __restrict__ out,
                   int R, int C) {
    __shared__ float tile[32][33];
    const int c0 = blockIdx.x * 32;
    const int r0 = blockIdx.y * 32;
    for (int j = threadIdx.y; j < 32; j += 8)
        tile[j][threadIdx.x] = in[(size_t)(r0 + j) * C + c0 + threadIdx.x];
    __syncthreads();
    for (int j = threadIdx.y; j < 32; j += 8)
        out[(size_t)(c0 + j) * R + r0 + threadIdx.x] = f2bf(tile[threadIdx.x][j]);
}

// ---------------------------------------------------------------------------
// LayerNorm over last dim (1024) -> bf16. One 256-thread block per row.
// ---------------------------------------------------------------------------
__global__ __launch_bounds__(256)
void layernorm_bf16(const float* __restrict__ x, const float* __restrict__ g,
                    const float* __restrict__ b, unsigned short* __restrict__ out) {
    __shared__ float rs1[256];
    __shared__ float rs2[256];
    const int row = blockIdx.x;
    const int tid = threadIdx.x;
    const float4* xr = (const float4*)(x + (size_t)row * DIM);
    float4 v = xr[tid];
    rs1[tid] = v.x + v.y + v.z + v.w;
    rs2[tid] = v.x * v.x + v.y * v.y + v.z * v.z + v.w * v.w;
    __syncthreads();
    for (int off = 128; off > 0; off >>= 1) {
        if (tid < off) { rs1[tid] += rs1[tid + off]; rs2[tid] += rs2[tid + off]; }
        __syncthreads();
    }
    const float mu  = rs1[0] * (1.0f / DIM);
    const float var = rs2[0] * (1.0f / DIM) - mu * mu;
    const float rst = rsqrtf(var + 1e-5f);
    const int c = tid * 4;
    unsigned short* o = out + (size_t)row * DIM + c;
    o[0] = f2bf((v.x - mu) * rst * g[c + 0] + b[c + 0]);
    o[1] = f2bf((v.y - mu) * rst * g[c + 1] + b[c + 1]);
    o[2] = f2bf((v.z - mu) * rst * g[c + 2] + b[c + 2]);
    o[3] = f2bf((v.w - mu) * rst * g[c + 3] + b[c + 3]);
}

// ---------------------------------------------------------------------------
// Row softmax over SEQ=2048 fp32 scores -> bf16 probs. 256 thr x 8 elems.
// ---------------------------------------------------------------------------
__global__ __launch_bounds__(256)
void softmax_bf16(const float* __restrict__ s, unsigned short* __restrict__ p) {
    __shared__ float red[256];
    const size_t row = blockIdx.x;
    const float* sr = s + row * SEQ;
    unsigned short* pr = p + row * SEQ;
    const int tid = threadIdx.x;
    float v[8];
    float m = -1e30f;
#pragma unroll
    for (int i = 0; i < 8; i++) { v[i] = sr[tid * 8 + i]; m = fmaxf(m, v[i]); }
    red[tid] = m;
    __syncthreads();
    for (int off = 128; off > 0; off >>= 1) {
        if (tid < off) red[tid] = fmaxf(red[tid], red[tid + off]);
        __syncthreads();
    }
    m = red[0];
    __syncthreads();
    float sum = 0.f;
#pragma unroll
    for (int i = 0; i < 8; i++) { v[i] = __expf(v[i] - m); sum += v[i]; }
    red[tid] = sum;
    __syncthreads();
    for (int off = 128; off > 0; off >>= 1) {
        if (tid < off) red[tid] += red[tid + off];
        __syncthreads();
    }
    const float inv = 1.0f / red[0];
#pragma unroll
    for (int i = 0; i < 8; i++) pr[tid * 8 + i] = f2bf(v[i] * inv);
}

// ---------------------------------------------------------------------------
// bf16 WMMA GEMM, f32 accumulate.  C[M,N] = A[M,K] @ Bt^T, Bt is [N][K].
// Tile: 128x128 per 256-thread block (8 wave32s; each wave 32x64 = 2x4 frags).
// M,N multiples of 128; K multiple of 32 (true for all calls here).
// Double-buffered LDS, async global->LDS staging.
// Epilogues: 0 = store bf16 row-major
//            1 = store fp32 * scale
//            2 = store fp32 (acc + bias[n] + res[m,n])
//            3 = store bf16 relu(acc + bias[n])
//            4 = store bf16 transposed per sub-batch of tRows rows:
//                Out[(row/tRows)*N*tRows + col*tRows + row%tRows]
// ---------------------------------------------------------------------------
template <int EPI>
__global__ __launch_bounds__(256)
void gemm_bf16(const unsigned short* __restrict__ A,
               const unsigned short* __restrict__ Bt,
               void* __restrict__ Out,
               const float* __restrict__ bias,
               const float* __restrict__ Res,
               float scale, int M, int N, int K, int tRows,
               long long aBS, long long bBS, long long oBS, long long rBS) {
    __shared__ __align__(16) unsigned short ldsA[2][128 * LDSK];
    __shared__ __align__(16) unsigned short ldsB[2][128 * LDSK];

    const int z = blockIdx.z;
    A  += (size_t)z * aBS;
    Bt += (size_t)z * bBS;

    const int tid   = threadIdx.x;
    const int lane  = tid & 31;
    const int wave  = tid >> 5;
    const int waveM = wave >> 1;   // 0..3 -> 32-row slabs
    const int waveN = wave & 1;    // 0..1 -> 64-col slabs
    const int half  = lane >> 4;
    const int l16   = lane & 15;

    const int mBase = blockIdx.y * 128;
    const int nBase = blockIdx.x * 128;

    const int sRow  = tid >> 1;    // 0..127 staging row
    const int sHalf = tid & 1;     // which 16-elem half of the 32-wide K slab

    const unsigned short* gA = A  + (size_t)(mBase + sRow) * K + sHalf * 16;
    const unsigned short* gB = Bt + (size_t)(nBase + sRow) * K + sHalf * 16;
    unsigned short* lA0 = &ldsA[0][(size_t)sRow * LDSK + sHalf * 16];
    unsigned short* lB0 = &ldsB[0][(size_t)sRow * LDSK + sHalf * 16];
    const size_t bufStride = 128 * LDSK;

    v8f acc[2][4] = {};
    union Frag { v16bf v; uint32_t u[8]; };

    const int nk = K >> 5;

    // prologue: stage slab 0 into buffer 0 (4 async ops / thread)
    stage32B(lA0, gA);
    stage32B(lB0, gB);

    for (int i = 0; i < nk; ++i) {
        const int cur = i & 1;
        if (i + 1 < nk) {
            const int nxt = 1 - cur;
            stage32B(lA0 + nxt * bufStride, gA + (size_t)(i + 1) * 32);
            stage32B(lB0 + nxt * bufStride, gB + (size_t)(i + 1) * 32);
            wait_async_4();          // current slab's 4 ops complete
        } else {
            wait_async_0();
        }
        __syncthreads();

        // ---- load fragments from current buffer ----
        // A 16x32 bf16: lane(l16,half): row m=l16;
        //   u[j] = K {2j,2j+1} + half*8 ; u[4+j] = 16 + same
        Frag af[2];
#pragma unroll
        for (int fm = 0; fm < 2; fm++) {
            const uint32_t* r = (const uint32_t*)
                (&ldsA[cur][(size_t)(waveM * 32 + fm * 16 + l16) * LDSK]);
#pragma unroll
            for (int j = 0; j < 4; j++) {
                af[fm].u[j]     = r[half * 4 + j];
                af[fm].u[4 + j] = r[8 + half * 4 + j];
            }
        }
        // B 32x16 bf16: lane(l16,half): col n=l16; u[j] = K {2j,2j+1} + half*8 ; u[4+j] = 16 + same
        Frag bfr[4];
#pragma unroll
        for (int fn = 0; fn < 4; fn++) {
            const uint32_t* r = (const uint32_t*)
                (&ldsB[cur][(size_t)(waveN * 64 + fn * 16 + l16) * LDSK]);
#pragma unroll
            for (int j = 0; j < 4; j++) {
                bfr[fn].u[j]     = r[half * 4 + j];
                bfr[fn].u[4 + j] = r[8 + half * 4 + j];
            }
        }

        // ---- 8 WMMAs per wave per K-step ----
#pragma unroll
        for (int fm = 0; fm < 2; fm++)
#pragma unroll
            for (int fn = 0; fn < 4; fn++)
                acc[fm][fn] = __builtin_amdgcn_wmma_f32_16x16x32_bf16(
                    false, af[fm].v, false, bfr[fn].v,
                    (short)0, acc[fm][fn], false, false);

        __syncthreads();
    }

    // ---- epilogue: C/D layout: VGPR r -> M = 8*half + r, N = l16 ----
#pragma unroll
    for (int fm = 0; fm < 2; fm++) {
#pragma unroll
        for (int fn = 0; fn < 4; fn++) {
            const int col  = nBase + waveN * 64 + fn * 16 + l16;
            const int row0 = mBase + waveM * 32 + fm * 16 + half * 8;
            if (EPI == 4) {
                // transposed bf16: lane's 8 rows are contiguous -> one b128 store
                unsigned short tmp[8];
#pragma unroll
                for (int r = 0; r < 8; r++) tmp[r] = f2bf(acc[fm][fn][r]);
                const int bz = row0 / tRows;
                const int rb = row0 % tRows;
                const size_t off = ((size_t)bz * N + col) * (size_t)tRows + rb;
                *(uint4*)((unsigned short*)Out + off) = *(const uint4*)tmp;
            } else {
#pragma unroll
                for (int r = 0; r < 8; r++) {
                    const int row = row0 + r;
                    const size_t oi = (size_t)row * N + col;
                    const float v = acc[fm][fn][r];
                    if (EPI == 0) {
                        ((unsigned short*)Out)[(size_t)z * oBS + oi] = f2bf(v);
                    } else if (EPI == 1) {
                        ((float*)Out)[(size_t)z * oBS + oi] = v * scale;
                    } else if (EPI == 2) {
                        ((float*)Out)[(size_t)z * oBS + oi] =
                            v + bias[col] + Res[(size_t)z * rBS + oi];
                    } else {
                        float t = v + bias[col];
                        t = t > 0.f ? t : 0.f;
                        ((unsigned short*)Out)[(size_t)z * oBS + oi] = f2bf(t);
                    }
                }
            }
        }
    }
}

// ---------------------------------------------------------------------------
// Workspace layout (bytes); needs ~272 MB scratch.
// ---------------------------------------------------------------------------
#define OFF_WQ   ((size_t)0)                      // bf16 Wq^T  [DIM][DIM]  2 MB
#define OFF_WK   ((size_t)(2u   * 1024 * 1024))
#define OFF_WV   ((size_t)(4u   * 1024 * 1024))
#define OFF_WO   ((size_t)(6u   * 1024 * 1024))
#define OFF_W1   ((size_t)(8u   * 1024 * 1024))   // W1^T [FF][DIM] 4 MB
#define OFF_W2   ((size_t)(12u  * 1024 * 1024))   // W2^T [DIM][FF] 4 MB
#define OFF_LN1  ((size_t)(16u  * 1024 * 1024))   // bf16, 16 MB
#define OFF_Q    ((size_t)(32u  * 1024 * 1024))   // bf16, 16 MB
#define OFF_K    ((size_t)(48u  * 1024 * 1024))   // bf16, 16 MB
#define OFF_V    ((size_t)(64u  * 1024 * 1024))   // bf16 V^T per batch, 16 MB
#define OFF_S    ((size_t)(80u  * 1024 * 1024))   // fp32 scores, 64 MB
#define OFF_P    ((size_t)(144u * 1024 * 1024))   // bf16 probs,  32 MB
#define OFF_ATT  ((size_t)(176u * 1024 * 1024))   // bf16 attn,   16 MB
#define OFF_X1   ((size_t)(192u * 1024 * 1024))   // fp32 x1,     32 MB
#define OFF_LN2  ((size_t)(224u * 1024 * 1024))   // bf16 ln2,    16 MB
#define OFF_H    ((size_t)(240u * 1024 * 1024))   // bf16 ffn hid,32 MB

extern "C" void kernel_launch(void* const* d_in, const int* in_sizes, int n_in,
                              void* d_out, int out_size, void* d_ws, size_t ws_size,
                              hipStream_t stream) {
    const float* x     = (const float*)d_in[0];
    const float* ln1_g = (const float*)d_in[1];
    const float* ln1_b = (const float*)d_in[2];
    const float* Wq    = (const float*)d_in[3];
    const float* Wk    = (const float*)d_in[4];
    const float* Wv    = (const float*)d_in[5];
    const float* Wo    = (const float*)d_in[6];
    const float* bo    = (const float*)d_in[7];
    const float* ln2_g = (const float*)d_in[8];
    const float* ln2_b = (const float*)d_in[9];
    const float* W1    = (const float*)d_in[10];
    const float* b1    = (const float*)d_in[11];
    const float* W2    = (const float*)d_in[12];
    const float* b2    = (const float*)d_in[13];

    char* ws = (char*)d_ws;
    unsigned short* wqT  = (unsigned short*)(ws + OFF_WQ);
    unsigned short* wkT  = (unsigned short*)(ws + OFF_WK);
    unsigned short* wvT  = (unsigned short*)(ws + OFF_WV);
    unsigned short* woT  = (unsigned short*)(ws + OFF_WO);
    unsigned short* w1T  = (unsigned short*)(ws + OFF_W1);
    unsigned short* w2T  = (unsigned short*)(ws + OFF_W2);
    unsigned short* ln1o = (unsigned short*)(ws + OFF_LN1);
    unsigned short* Qb   = (unsigned short*)(ws + OFF_Q);
    unsigned short* Kb   = (unsigned short*)(ws + OFF_K);
    unsigned short* Vt   = (unsigned short*)(ws + OFF_V);
    float*          Sc   = (float*)         (ws + OFF_S);
    unsigned short* Pb   = (unsigned short*)(ws + OFF_P);
    unsigned short* Att  = (unsigned short*)(ws + OFF_ATT);
    float*          X1   = (float*)         (ws + OFF_X1);
    unsigned short* ln2o = (unsigned short*)(ws + OFF_LN2);
    unsigned short* Hb   = (unsigned short*)(ws + OFF_H);
    float*          Out  = (float*)d_out;

    dim3 tb(32, 8);
    // 1) transpose+convert all weights: W[K][N] fp32 -> W^T[N][K] bf16
    transpose_cvt<<<dim3(DIM / 32, DIM / 32), tb, 0, stream>>>(Wq, wqT, DIM, DIM);
    transpose_cvt<<<dim3(DIM / 32, DIM / 32), tb, 0, stream>>>(Wk, wkT, DIM, DIM);
    transpose_cvt<<<dim3(DIM / 32, DIM / 32), tb, 0, stream>>>(Wv, wvT, DIM, DIM);
    transpose_cvt<<<dim3(DIM / 32, DIM / 32), tb, 0, stream>>>(Wo, woT, DIM, DIM);
    transpose_cvt<<<dim3(FF  / 32, DIM / 32), tb, 0, stream>>>(W1, w1T, DIM, FF);
    transpose_cvt<<<dim3(DIM / 32, FF  / 32), tb, 0, stream>>>(W2, w2T, FF, DIM);

    // 2) LN1
    layernorm_bf16<<<ROWS, 256, 0, stream>>>(x, ln1_g, ln1_b, ln1o);

    // 3) Q, K, V projections: [8192,1024] @ [1024,1024]; V stored transposed
    dim3 gQKV(DIM / 128, ROWS / 128, 1);
    gemm_bf16<0><<<gQKV, 256, 0, stream>>>(ln1o, wqT, Qb, nullptr, nullptr,
        1.f, ROWS, DIM, DIM, SEQ, 0, 0, 0, 0);
    gemm_bf16<0><<<gQKV, 256, 0, stream>>>(ln1o, wkT, Kb, nullptr, nullptr,
        1.f, ROWS, DIM, DIM, SEQ, 0, 0, 0, 0);
    gemm_bf16<4><<<gQKV, 256, 0, stream>>>(ln1o, wvT, Vt, nullptr, nullptr,
        1.f, ROWS, DIM, DIM, SEQ, 0, 0, 0, 0);

    // 4) scores = (Q @ K^T) / sqrt(d), batched (K is already [N][K] form)
    dim3 gSc(SEQ / 128, SEQ / 128, BATCH);
    gemm_bf16<1><<<gSc, 256, 0, stream>>>(Qb, Kb, Sc, nullptr, nullptr,
        0.03125f, SEQ, SEQ, DIM, SEQ,
        (long long)SEQ * DIM, (long long)SEQ * DIM, (long long)SEQ * SEQ, 0);

    // 5) softmax rows
    softmax_bf16<<<ROWS, 256, 0, stream>>>(Sc, Pb);

    // 6) attn_out = P @ V, batched (B^T source = V^T [DIM][SEQ] per batch)
    dim3 gAV(DIM / 128, SEQ / 128, BATCH);
    gemm_bf16<0><<<gAV, 256, 0, stream>>>(Pb, Vt, Att, nullptr, nullptr,
        1.f, SEQ, DIM, SEQ, SEQ,
        (long long)SEQ * SEQ, (long long)DIM * SEQ, (long long)SEQ * DIM, 0);

    // 7) x1 = attn_out @ Wo + bo + x
    gemm_bf16<2><<<gQKV, 256, 0, stream>>>(Att, woT, X1, bo, x,
        1.f, ROWS, DIM, DIM, SEQ, 0, 0, 0, 0);

    // 8) LN2
    layernorm_bf16<<<ROWS, 256, 0, stream>>>(X1, ln2_g, ln2_b, ln2o);

    // 9) h = relu(ln2 @ W1 + b1)
    dim3 gF1(FF / 128, ROWS / 128, 1);
    gemm_bf16<3><<<gF1, 256, 0, stream>>>(ln2o, w1T, Hb, b1, nullptr,
        1.f, ROWS, FF, DIM, SEQ, 0, 0, 0, 0);

    // 10) out = h @ W2 + b2 + x1
    gemm_bf16<2><<<gQKV, 256, 0, stream>>>(Hb, w2T, Out, b2, X1,
        1.f, ROWS, DIM, FF, SEQ, 0, 0, 0, 0);
}
